// ScaledDotProductAttention_80229989089584
// MI455X (gfx1250) — matrix operand order repaired and checked by reading, not compiled, not run on hardware
//
#include <hip/hip_runtime.h>
#include <hip/hip_bf16.h>

typedef __attribute__((ext_vector_type(16))) _Float16 v16h;
typedef __attribute__((ext_vector_type(8)))  _Float16 v8h;
typedef __attribute__((ext_vector_type(8)))  float    v8f;
typedef __fp16 v2hf __attribute__((ext_vector_type(2)));
typedef __fp16 v4hf __attribute__((ext_vector_type(4)));
typedef __fp16 v8hf __attribute__((ext_vector_type(8)));

#define WAVES_PER_BLOCK 4
#define KEYS_PER_ITER   32
#define DK              64
#define QROWS_BLOCK     64   // WAVES_PER_BLOCK * 16
#define LOG2E           1.4426950408889634f

// 16 contiguous f16 (32B) -> one B-fragment per-lane run: two ds_load_b128
__device__ __forceinline__ v16h ld_frag16(const _Float16* p) {
    v8h lo = *(const v8h*)(p);
    v8h hh = *(const v8h*)(p + 8);
    return __builtin_shufflevector(lo, hh, 0,1,2,3,4,5,6,7,8,9,10,11,12,13,14,15);
}

__device__ __forceinline__ v8f wmma_f16(v16h a, v16h b, v8f c) {
    return __builtin_amdgcn_wmma_f32_16x16x32_f16(
        /*neg_a=*/false, a, /*neg_b=*/false, b,
        /*c_mod=*/(short)0, c, /*reuse_a=*/false, /*reuse_b=*/false);
}

// pack 8 f32 -> 8 f16 with packed converts (v_cvt_pk_rtz_f16_f32), optional scale
__device__ __forceinline__ v8h cvt8(const float4 a, const float4 b, const float s) {
    v2hf p0 = __builtin_amdgcn_cvt_pkrtz(a.x * s, a.y * s);
    v2hf p1 = __builtin_amdgcn_cvt_pkrtz(a.z * s, a.w * s);
    v2hf p2 = __builtin_amdgcn_cvt_pkrtz(b.x * s, b.y * s);
    v2hf p3 = __builtin_amdgcn_cvt_pkrtz(b.z * s, b.w * s);
    v4hf q0 = __builtin_shufflevector(p0, p1, 0,1,2,3);
    v4hf q1 = __builtin_shufflevector(p2, p3, 0,1,2,3);
    v8hf r  = __builtin_shufflevector(q0, q1, 0,1,2,3,4,5,6,7);
    return __builtin_bit_cast(v8h, r);
}

// CDNA5 async copy: 16B per lane, global -> LDS, ASYNCcnt-tracked, non-temporal
__device__ __forceinline__ void async_b128_to_lds_nt(void* ldsp, const void* gp) {
    const unsigned lds_off = (unsigned)(unsigned long long)(uintptr_t)ldsp;
    asm volatile("global_load_async_to_lds_b128 %0, %1, off th:TH_LOAD_NT"
                 :: "v"(lds_off), "v"(gp) : "memory");
}

// raw v_max_num_f32 (no canonicalize ops around it)
__device__ __forceinline__ float vmaxf(float a, float b) {
    float d;
    asm("v_max_num_f32_e32 %0, %1, %2" : "=v"(d) : "v"(a), "v"(b));
    return d;
}

// fused DPP row_ror butterflies within 16-lane DPP rows: 1 VALU per step
__device__ __forceinline__ float rowmax16(float v) {
    asm("v_max_num_f32_dpp %0, %1, %1 row_ror:1 row_mask:0xf bank_mask:0xf bound_ctrl:1" : "=v"(v) : "v"(v));
    asm("v_max_num_f32_dpp %0, %1, %1 row_ror:2 row_mask:0xf bank_mask:0xf bound_ctrl:1" : "=v"(v) : "v"(v));
    asm("v_max_num_f32_dpp %0, %1, %1 row_ror:4 row_mask:0xf bank_mask:0xf bound_ctrl:1" : "=v"(v) : "v"(v));
    asm("v_max_num_f32_dpp %0, %1, %1 row_ror:8 row_mask:0xf bank_mask:0xf bound_ctrl:1" : "=v"(v) : "v"(v));
    return v;
}
__device__ __forceinline__ float rowsum16(float v) {
    asm("v_add_f32_dpp %0, %1, %1 row_ror:1 row_mask:0xf bank_mask:0xf bound_ctrl:1" : "=v"(v) : "v"(v));
    asm("v_add_f32_dpp %0, %1, %1 row_ror:2 row_mask:0xf bank_mask:0xf bound_ctrl:1" : "=v"(v) : "v"(v));
    asm("v_add_f32_dpp %0, %1, %1 row_ror:4 row_mask:0xf bank_mask:0xf bound_ctrl:1" : "=v"(v) : "v"(v));
    asm("v_add_f32_dpp %0, %1, %1 row_ror:8 row_mask:0xf bank_mask:0xf bound_ctrl:1" : "=v"(v) : "v"(v));
    return v;
}

__global__ __launch_bounds__(128, 2)
void fa_fwd_kernel(const float* __restrict__ Q,
                   const float* __restrict__ K,
                   const float* __restrict__ V,
                   const unsigned char* __restrict__ Mask,  // bool, True => masked
                   float* __restrict__ Out,
                   int S)
{
    __shared__ __align__(16) _Float16 Klds[KEYS_PER_ITER * DK];                 // [key][d]   4 KB
    __shared__ __align__(16) _Float16 Vtld[DK * KEYS_PER_ITER];                 // [d][pos]   4 KB
    __shared__ __align__(16) _Float16 Plds[WAVES_PER_BLOCK][16 * KEYS_PER_ITER];// per-wave P 4 KB
    __shared__ __align__(16) unsigned char Mlds[QROWS_BLOCK * KEYS_PER_ITER];   // mask tile  2 KB

    const int tid   = threadIdx.x;
    const int lane  = tid & 31;
    const int wave  = tid >> 5;
    const int ln16  = lane & 15;
    const bool hi   = lane >= 16;
    const int rbase = hi ? 8 : 0;   // C-layout: VGPR r -> row r (lanes 0-15) / r+8 (lanes 16-31)
    const int kbase = hi ? 8 : 0;   // table-order K-run base; unused (A fragments use koff too)
    const int koff  = hi ? 16 : 0;  // K-run base for both A and B fragments (contiguous k = 16h + s)

    const int qtiles = S / QROWS_BLOCK;
    const int qtile  = blockIdx.x % qtiles;
    const int bh     = blockIdx.x / qtiles;
    const int q0blk  = qtile * QROWS_BLOCK;
    const int q0     = q0blk + wave * 16;

    const float* __restrict__ Qbh = Q + (size_t)bh * S * DK;
    const float* __restrict__ Kbh = K + (size_t)bh * S * DK;
    const float* __restrict__ Vbh = V + (size_t)bh * S * DK;
    const unsigned char* __restrict__ Mbh = Mask + (size_t)bh * S * S;
    float* __restrict__ Obh = Out + (size_t)bh * S * DK;

    // ---- Q once, pre-scaled by (1/sqrt(64))*log2(e): base-2 softmax domain ----
    v16h qa[2];
#pragma unroll
    for (int f = 0; f < 2; ++f) {
        const float* qrow = Qbh + (size_t)(q0 + ln16) * DK + f * 32;
        v8h run[2];
#pragma unroll
        for (int rr = 0; rr < 2; ++rr) {
            const float4 x0 = *(const float4*)(qrow + koff + rr * 8);
            const float4 x1 = *(const float4*)(qrow + koff + rr * 8 + 4);
            run[rr] = cvt8(x0, x1, 0.125f * LOG2E);
        }
        qa[f] = __builtin_shufflevector(run[0], run[1], 0,1,2,3,4,5,6,7,8,9,10,11,12,13,14,15);
    }

    // ---- online-softmax state: 8 rows per lane ----
    float m_run[8], l_run[8];
    v8f acc[4];
#pragma unroll
    for (int r = 0; r < 8; ++r) { m_run[r] = -__builtin_inff(); l_run[r] = 0.0f; }
#pragma unroll
    for (int t = 0; t < 4; ++t) acc[t] = (v8f){0.f,0.f,0.f,0.f,0.f,0.f,0.f,0.f};

    // cooperative staging induction pointers
    const int ckey = tid >> 2;          // 0..31
    const int cd0  = (tid & 3) * 16;    // 0,16,32,48
    // interleaved key permutation (pos 2j <-> key j, pos 2j+1 <-> key 16+j);
    // applied identically to P's k-dim and V's rows -> PV sum unchanged
    const int vpos = (ckey < 16) ? (2 * ckey) : (2 * (ckey - 16) + 1);
    const float* kp = Kbh + (size_t)ckey * DK + cd0;
    const float* vp = Vbh + (size_t)ckey * DK + cd0;
    const int mrow = tid >> 1;          // 0..63
    const int mseg = (tid & 1) * 16;    // 0 or 16
    const unsigned char* mp = Mbh + (size_t)(q0blk + mrow) * S + mseg;
    unsigned char* mlp = &Mlds[mrow * KEYS_PER_ITER + mseg];

    for (int j0 = 0; j0 < S; j0 += KEYS_PER_ITER) {
        __syncthreads();   // previous iteration's LDS consumers are done

        // mask tile: one coalesced non-temporal 16B async copy per thread
        async_b128_to_lds_nt(mlp, mp);

        {   // K (row-major f16) and V (transposed + permuted f16) staging
            const float4 k0 = *(const float4*)(kp + 0),  k1 = *(const float4*)(kp + 4);
            const float4 k2 = *(const float4*)(kp + 8),  k3 = *(const float4*)(kp + 12);
            const float4 w0 = *(const float4*)(vp + 0),  w1 = *(const float4*)(vp + 4);
            const float4 w2 = *(const float4*)(vp + 8),  w3 = *(const float4*)(vp + 12);
            if (j0 + KEYS_PER_ITER < S) {   // global_prefetch_b8 for next tiles
                __builtin_prefetch(kp + (size_t)KEYS_PER_ITER * DK, 0, 1);
                __builtin_prefetch(vp + (size_t)KEYS_PER_ITER * DK, 0, 1);
                __builtin_prefetch(mp + KEYS_PER_ITER, 0, 1);
            }
            _Float16* kd = &Klds[ckey * DK + cd0];
            *(v8h*)(kd)     = cvt8(k0, k1, 1.0f);
            *(v8h*)(kd + 8) = cvt8(k2, k3, 1.0f);
            const float vv[16] = {w0.x,w0.y,w0.z,w0.w, w1.x,w1.y,w1.z,w1.w,
                                  w2.x,w2.y,w2.z,w2.w, w3.x,w3.y,w3.z,w3.w};
#pragma unroll
            for (int i = 0; i < 16; ++i)
                Vtld[(cd0 + i) * KEYS_PER_ITER + vpos] = (_Float16)vv[i];
        }
        asm volatile("s_wait_asynccnt 0x0" ::: "memory");
        __syncthreads();   // tiles ready

        // ---- C-init = additive mask bias (0 or -1e9*log2e) folded into WMMA ----
        v8f c0, c1;
#pragma unroll
        for (int r = 0; r < 8; ++r) {
            const int wrow = wave * 16 + rbase + r;
            c0[r] = (-1e9f * LOG2E) * (float)Mlds[wrow * KEYS_PER_ITER + ln16];
            c1[r] = (-1e9f * LOG2E) * (float)Mlds[wrow * KEYS_PER_ITER + 16 + ln16];
        }

        // ---- scores (base-2 domain): S = (Q*scale*log2e) · K^T + bias ----
#pragma unroll
        for (int f = 0; f < 2; ++f) {
            const v16h b0 = ld_frag16(&Klds[(ln16)      * DK + f * 32 + koff]);
            const v16h b1 = ld_frag16(&Klds[(16 + ln16) * DK + f * 32 + koff]);
            c0 = wmma_f16(qa[f], b0, c0);
            c1 = wmma_f16(qa[f], b1, c1);
        }

        // ---- online softmax: fused DPP butterflies + raw exp2 ----
#pragma unroll
        for (int r = 0; r < 8; ++r) {
            const float mx    = rowmax16(vmaxf(c0[r], c1[r]));
            const float mnew  = vmaxf(m_run[r], mx);
            const float alpha = __builtin_amdgcn_exp2f(m_run[r] - mnew);
            const float p0 = __builtin_amdgcn_exp2f(c0[r] - mnew);
            const float p1 = __builtin_amdgcn_exp2f(c1[r] - mnew);
            const float rs = rowsum16(p0 + p1);
            l_run[r] = l_run[r] * alpha + rs;
            m_run[r] = mnew;
#pragma unroll
            for (int t = 0; t < 4; ++t) acc[t][r] *= alpha;
            // packed store: (p0,p1) -> interleaved positions (2*ln16, 2*ln16+1)
            const v2hf ph = __builtin_amdgcn_cvt_pkrtz(p0, p1);
            *(unsigned*)&Plds[wave][(rbase + r) * KEYS_PER_ITER + 2 * ln16] =
                __builtin_bit_cast(unsigned, ph);
        }
        // same-wave LDS store->load (LDS is in-order per wave)
        asm volatile("s_wait_dscnt 0x0" ::: "memory");

        // ---- re-layout P (C-layout -> A-layout 16x32) via per-wave LDS ----
        v16h pa;
        {
            const _Float16* base = &Plds[wave][ln16 * KEYS_PER_ITER];
            v8h lo = *(const v8h*)(base + koff);
            v8h hh = *(const v8h*)(base + koff + 8);
            pa = __builtin_shufflevector(lo, hh, 0,1,2,3,4,5,6,7,8,9,10,11,12,13,14,15);
        }

        // ---- O += P · V : 4 output tiles, contiguous B-frags from permuted V ----
#pragma unroll
        for (int t = 0; t < 4; ++t) {
            const v16h vb = ld_frag16(&Vtld[(t * 16 + ln16) * KEYS_PER_ITER + koff]);
            acc[t] = wmma_f16(pa, vb, acc[t]);
        }

        kp += (size_t)KEYS_PER_ITER * DK;
        vp += (size_t)KEYS_PER_ITER * DK;
        mp += KEYS_PER_ITER;
    }

    // ---- epilogue: divide by row sums, store f32 ----
#pragma unroll
    for (int r = 0; r < 8; ++r) {
        const float invl = 1.0f / l_run[r];
        float* orow = Obh + (size_t)(q0 + rbase + r) * DK;
#pragma unroll
        for (int t = 0; t < 4; ++t)
            orow[t * 16 + ln16] = acc[t][r] * invl;
    }
}

extern "C" void kernel_launch(void* const* d_in, const int* in_sizes, int n_in,
                              void* d_out, int out_size, void* d_ws, size_t ws_size,
                              hipStream_t stream) {
    const float*         Q = (const float*)d_in[0];
    const float*         K = (const float*)d_in[1];
    const float*         V = (const float*)d_in[2];
    const unsigned char* M = (const unsigned char*)d_in[3]; // jax bool -> 1 byte
    float*               O = (float*)d_out;

    const int B = 2, H = 16, S = 2048; // fixed by the reference module
    dim3 grid(B * H * (S / QROWS_BLOCK));
    dim3 block(WAVES_PER_BLOCK * 32);
    hipLaunchKernelGGL(fa_fwd_kernel, grid, block, 0, stream, Q, K, V, M, O, S);
}
